// CrossAttention_44693429682634
// MI455X (gfx1250) — matrix operand order repaired and checked by reading, not compiled, not run on hardware
//
#include <hip/hip_runtime.h>

// ---------------------------------------------------------------------------
// CDNA5 (gfx1250) fused cross-attention: bf16 WMMA everywhere + TDM staging.
//   pass 1: Qp = (q@Wq + bq) * 1/sqrt(D)   [bf16, row-major]
//           Kp = (k@Wk + bk)               [bf16, row-major]
//           Vt = (v@Wv + bv)               [bf16, TRANSPOSED per batch: [B][D][S]]
//   pass 2: flash attention (online softmax), AO = softmax(Qp Kp^T) Vt  [bf16]
//   pass 3: out = AO@Wo + bo               [fp32]
// Workspace: 4 x 64MB bf16 buffers = 256 MB.
// ---------------------------------------------------------------------------

typedef __attribute__((ext_vector_type(16))) __bf16       v16bf;
typedef __attribute__((ext_vector_type(8)))  __bf16       v8bf;
typedef __attribute__((ext_vector_type(8)))  float        v8f;
typedef __attribute__((ext_vector_type(4)))  float        f32x4;
typedef __attribute__((ext_vector_type(4)))  unsigned int u32x4;
typedef __attribute__((ext_vector_type(4)))  unsigned int v4u;
typedef __attribute__((ext_vector_type(4)))  int          v4i;
typedef __attribute__((ext_vector_type(8)))  int          v8i;

union Frag16 { v16bf v; u32x4 q[2]; };
union Pack8  { v8bf  v; u32x4 q;    };

constexpr int D_MODEL = 1024;
constexpr int SEQ     = 2048;
constexpr int BATCH   = 16;
constexpr int M_TOTAL = BATCH * SEQ;   // 32768 flattened rows

// ---- Tensor Data Mover support (guarded: fallback = manual copy) ----------
#if defined(__has_builtin)
#if __has_builtin(__builtin_amdgcn_tensor_load_to_lds) && \
    __has_builtin(__builtin_amdgcn_s_wait_tensorcnt)
#define CDNA5_HAS_TDM 1
#endif
#endif

#ifdef CDNA5_HAS_TDM
// 2-D bf16 tile load, global -> LDS, per CDNA5 ISA.
// D# group0: count=1, lds_addr, 57-bit global addr, type=2.
// D# group1: data_size=2B, tensor dims (large), tile_dim0 x tile_dim1, stride.
static __device__ __forceinline__ void tdm_load_2d_bf16(unsigned lds_addr,
                                                        const void* gptr,
                                                        int tile_elems_x,
                                                        int tile_rows,
                                                        int stride_elems) {
  const unsigned long long ga = (unsigned long long)(size_t)gptr;
  v4u g0;
  g0[0] = 1u;                                            // count=1, user mode
  g0[1] = lds_addr;                                      // LDS byte address
  g0[2] = (unsigned)(ga & 0xFFFFFFFFu);                  // global addr [31:0]
  g0[3] = (unsigned)((ga >> 32) & 0x01FFFFFFu) | (2u << 30);  // [56:32]|type=2
  const unsigned TD = 1u << 20;   // generous tensor dims (tiles are in-bounds)
  v8i g1;
  g1[0] = (int)(1u << 16);                               // data_size = 1 (2B)
  g1[1] = (int)((TD & 0xFFFFu) << 16);                   // tensor_dim0 lo16
  g1[2] = (int)(((TD >> 16) & 0xFFFFu) | ((TD & 0xFFFFu) << 16)); // d0 hi|d1 lo
  g1[3] = (int)(((TD >> 16) & 0xFFFFu) |
                ((unsigned)tile_elems_x << 16));         // d1 hi | tile_dim0
  g1[4] = (int)((unsigned)tile_rows & 0xFFFFu);          // tile_dim1 (dim2=0)
  g1[5] = (int)(unsigned)stride_elems;                   // tensor_dim0_stride
  g1[6] = 0;
  g1[7] = 0;
  v4i z4 = {0, 0, 0, 0};
#if defined(__clang_major__) && __clang_major__ >= 23
  v8i z8 = {0, 0, 0, 0, 0, 0, 0, 0};
  __builtin_amdgcn_tensor_load_to_lds(g0, g1, z4, z4, z8, 0);
#else
  __builtin_amdgcn_tensor_load_to_lds(g0, g1, z4, z4, 0);
#endif
}
#endif  // CDNA5_HAS_TDM

static __device__ __forceinline__ v8f zero8() {
  v8f z;
#pragma unroll
  for (int i = 0; i < 8; ++i) z[i] = 0.0f;
  return z;
}

static __device__ __forceinline__ v8f wmma_bf16(v16bf a, v16bf b, v8f c) {
  // D = A(16x32 bf16) * B(32x16 bf16) + C(16x16 f32)
  return __builtin_amdgcn_wmma_f32_16x16x32_bf16(false, a, false, b,
                                                 (short)0, c, false, false);
}

// ---------------------------------------------------------------------------
// GEMM: out = alpha * (A[M,1024] @ W[1024,1024] + bias)
//   AT      : float (convert to bf16 in LDS) or __bf16 (TDM / copy)
//   OMODE 0 : bf16 row-major out
//   OMODE 1 : bf16 out transposed per batch -> [B][D_MODEL][SEQ]
//   OMODE 2 : fp32 row-major out
// 256 threads (8 waves), tile 128x128, k-step 32, double-buffered LDS.
// ---------------------------------------------------------------------------
template <typename AT, int OMODE>
__global__ __launch_bounds__(256) void gemm_proj(const AT* __restrict__ A,
                                                 const float* __restrict__ W,
                                                 const float* __restrict__ bias,
                                                 float alpha,
                                                 void* __restrict__ outp) {
  __shared__ __bf16 a_lds[2][128][32];    // A block (bf16), ping-pong
  __shared__ __bf16 wt_lds[2][128][32];   // W block TRANSPOSED: [n][k]

  const int t    = threadIdx.x;
  const int lane = t & 31;
  const int w    = t >> 5;
  const int wm   = w >> 1;   // 0..3 -> 32-row strip
  const int wn   = w & 1;    // 0..1 -> 64-col strip
  const int lm   = lane & 15;
  const int hh   = lane >> 4;

  const int n0 = blockIdx.x * 128;
  const int m0 = blockIdx.y * 128;

  v8f acc[2][4];
#pragma unroll
  for (int i = 0; i < 2; ++i)
#pragma unroll
    for (int j = 0; j < 4; ++j) acc[i][j] = zero8();

  const int ar = t >> 1;          // A stage: row 0..127
  const int ak = (t & 1) * 16;    //          k half
  const int wk = t >> 3;          // W stage: k row 0..31
  const int wc = (t & 7) * 16;    //          col group

  auto stage = [&](int k0, int buf) {
    // ---- stage A ----
    if constexpr (sizeof(AT) == 4) {
      const float* ap = (const float*)A + (size_t)(m0 + ar) * D_MODEL + k0 + ak;
#pragma unroll
      for (int j = 0; j < 4; ++j) {
        f32x4 f = *(const f32x4*)(ap + 4 * j);
        a_lds[buf][ar][ak + 4 * j + 0] = (__bf16)f.x;
        a_lds[buf][ar][ak + 4 * j + 1] = (__bf16)f.y;
        a_lds[buf][ar][ak + 4 * j + 2] = (__bf16)f.z;
        a_lds[buf][ar][ak + 4 * j + 3] = (__bf16)f.w;
      }
    } else {
#ifdef CDNA5_HAS_TDM
      // async 128x32 bf16 tile, exact row-major match of a_lds[buf]
      if (w == 0)
        tdm_load_2d_bf16((unsigned)(size_t)&a_lds[buf][0][0],
                         (const __bf16*)A + (size_t)m0 * D_MODEL + k0,
                         /*tile_x=*/32, /*rows=*/128, /*stride=*/D_MODEL);
#else
      const __bf16* ap =
          (const __bf16*)A + (size_t)(m0 + ar) * D_MODEL + k0 + ak;
      *(u32x4*)&a_lds[buf][ar][ak]     = *(const u32x4*)ap;
      *(u32x4*)&a_lds[buf][ar][ak + 8] = *(const u32x4*)(ap + 8);
#endif
    }
    // ---- stage W transposed (fp32 -> bf16) ----
    const float* wp = W + (size_t)(k0 + wk) * D_MODEL + n0 + wc;
#pragma unroll
    for (int j4 = 0; j4 < 4; ++j4) {
      f32x4 f = *(const f32x4*)(wp + 4 * j4);
      wt_lds[buf][wc + 4 * j4 + 0][wk] = (__bf16)f.x;
      wt_lds[buf][wc + 4 * j4 + 1][wk] = (__bf16)f.y;
      wt_lds[buf][wc + 4 * j4 + 2][wk] = (__bf16)f.z;
      wt_lds[buf][wc + 4 * j4 + 3][wk] = (__bf16)f.w;
    }
  };

  auto tdm_fence = [&]() {
#ifdef CDNA5_HAS_TDM
    if constexpr (sizeof(AT) == 2) {
      if (w == 0) __builtin_amdgcn_s_wait_tensorcnt(0);
    }
#endif
  };

  stage(0, 0);
  tdm_fence();
  __syncthreads();

  constexpr int KSTEPS = D_MODEL / 32;
  for (int ks = 0; ks < KSTEPS; ++ks) {
    const int cur = ks & 1;
    if (ks + 1 < KSTEPS) stage((ks + 1) * 32, cur ^ 1);  // overlap next stage

    Frag16 af[2];
#pragma unroll
    for (int mt = 0; mt < 2; ++mt) {
      const int row = wm * 32 + mt * 16 + lm;
      af[mt].q[0] = *(const u32x4*)&a_lds[cur][row][8 * hh];
      af[mt].q[1] = *(const u32x4*)&a_lds[cur][row][16 + 8 * hh];
    }
    Frag16 bfr[4];
#pragma unroll
    for (int nt = 0; nt < 4; ++nt) {
      const int col = wn * 64 + nt * 16 + lm;
      bfr[nt].q[0] = *(const u32x4*)&wt_lds[cur][col][8 * hh];
      bfr[nt].q[1] = *(const u32x4*)&wt_lds[cur][col][16 + 8 * hh];
    }
#pragma unroll
    for (int mt = 0; mt < 2; ++mt)
#pragma unroll
      for (int nt = 0; nt < 4; ++nt)
        acc[mt][nt] = wmma_bf16(af[mt].v, bfr[nt].v, acc[mt][nt]);

    tdm_fence();
    __syncthreads();
  }

  // ---- epilogue: bias, alpha, store ----
#pragma unroll
  for (int mt = 0; mt < 2; ++mt) {
    const int row_base = m0 + wm * 32 + mt * 16 + 8 * hh;  // rows row_base..+7
#pragma unroll
    for (int nt = 0; nt < 4; ++nt) {
      const int col = n0 + wn * 64 + nt * 16 + lm;
      const float bb = bias[col];
      v8f r = acc[mt][nt];
      if constexpr (OMODE == 2) {
        float* o = (float*)outp;
#pragma unroll
        for (int e = 0; e < 8; ++e)
          o[(size_t)(row_base + e) * D_MODEL + col] = alpha * (r[e] + bb);
      } else if constexpr (OMODE == 0) {
        __bf16* o = (__bf16*)outp;
#pragma unroll
        for (int e = 0; e < 8; ++e)
          o[(size_t)(row_base + e) * D_MODEL + col] =
              (__bf16)(alpha * (r[e] + bb));
      } else {
        // transposed per batch: contiguous 8x bf16 (16B) store along seq
        __bf16* o    = (__bf16*)outp;
        const int b  = row_base >> 11;     // / SEQ
        const int kv = row_base & 2047;    // % SEQ
        Pack8 p;
#pragma unroll
        for (int e = 0; e < 8; ++e) p.v[e] = (__bf16)(alpha * (r[e] + bb));
        *(u32x4*)(o + ((size_t)b * D_MODEL + col) * SEQ + kv) = p.q;
      }
    }
  }
}

// ---------------------------------------------------------------------------
// Fused flash attention. 1 workgroup = 16 query rows, 8 waves.
//   S phase : wave w computes S[:, w*16 .. w*16+16) over full D (Q in LDS)
//   softmax : cooperative online max/sum via LDS
//   PV phase: wave w owns output d-cols [w*128, w*128+128), V read from Vt
// grid = BATCH * SEQ/16 = 2048 workgroups. Per-batch K+V (8MB) is L2-resident.
// ---------------------------------------------------------------------------
__global__ __launch_bounds__(256) void attn_fused(const __bf16* __restrict__ Qp,
                                                  const __bf16* __restrict__ Kp,
                                                  const __bf16* __restrict__ Vt,
                                                  __bf16* __restrict__ AO) {
  __shared__ __bf16 q_lds[16][D_MODEL];  // 32 KB
  __shared__ float  s_buf[16][128];      // 8 KB
  __shared__ __bf16 p_buf[16][128];      // 4 KB
  __shared__ float  red[16][16];         // 1 KB
  __shared__ float  mrow[16], lrow[16], corr[16];

  const int t    = threadIdx.x;
  const int lane = t & 31;
  const int w    = t >> 5;
  const int lm   = lane & 15;
  const int hh   = lane >> 4;

  const int b  = blockIdx.x >> 7;          // batch
  const int q0 = (blockIdx.x & 127) << 4;  // q-tile start

  // load Q tile (16 x 1024 bf16 = 32KB) into LDS
#ifdef CDNA5_HAS_TDM
  if (w == 0) {
    tdm_load_2d_bf16((unsigned)(size_t)&q_lds[0][0],
                     Qp + (size_t)(b * SEQ + q0) * D_MODEL,
                     /*tile_x=*/D_MODEL, /*rows=*/16, /*stride=*/D_MODEL);
    __builtin_amdgcn_s_wait_tensorcnt(0);
  }
#else
  {
    const int qr = t >> 4, qc = (t & 15) * 64;
    const u32x4* src =
        (const u32x4*)(Qp + ((size_t)(b * SEQ + q0 + qr)) * D_MODEL + qc);
    u32x4* dst = (u32x4*)&q_lds[qr][qc];
#pragma unroll
    for (int j = 0; j < 8; ++j) dst[j] = src[j];
  }
#endif
  if (t < 16) { mrow[t] = -3.0e38f; lrow[t] = 0.0f; }
  __syncthreads();

  v8f o_acc[8];
#pragma unroll
  for (int i = 0; i < 8; ++i) o_acc[i] = zero8();

  const int r  = t >> 4;        // softmax: row
  const int c0 = (t & 15) * 8;  // softmax: col group

  for (int kv0 = 0; kv0 < SEQ; kv0 += 128) {
    // ---- S = Qs @ K^T ; wave w -> kv cols [w*16, w*16+16) ----
    v8f s_acc = zero8();
    const __bf16* krow =
        Kp + ((size_t)(b * SEQ + kv0 + w * 16 + lm)) * D_MODEL + 8 * hh;
    if (kv0 + 128 < SEQ)  // warm L0/WGP$ for next tile (global_prefetch_b8)
      __builtin_prefetch(krow + (size_t)128 * D_MODEL, 0, 1);
#pragma unroll 8
    for (int kd = 0; kd < D_MODEL; kd += 32) {
      Frag16 aq, bk;
      const __bf16* qp = &q_lds[lm][kd + 8 * hh];
      aq.q[0] = *(const u32x4*)qp;
      aq.q[1] = *(const u32x4*)(qp + 16);
      bk.q[0] = *(const u32x4*)(krow + kd);
      bk.q[1] = *(const u32x4*)(krow + kd + 16);
      s_acc = wmma_bf16(aq.v, bk.v, s_acc);
    }
#pragma unroll
    for (int e = 0; e < 8; ++e) s_buf[e + 8 * hh][w * 16 + lm] = s_acc[e];
    __syncthreads();

    // ---- online softmax over the 16x128 block ----
    float pm = -3.0e38f;
#pragma unroll
    for (int j = 0; j < 8; ++j) pm = fmaxf(pm, s_buf[r][c0 + j]);
    red[r][t & 15] = pm;
    __syncthreads();
    if (t < 16) {
      float m_ = mrow[t];
#pragma unroll
      for (int j = 0; j < 16; ++j) m_ = fmaxf(m_, red[t][j]);
      corr[t] = __expf(mrow[t] - m_);
      mrow[t] = m_;
    }
    __syncthreads();
    const float mnew = mrow[r];
    float psum = 0.0f;
#pragma unroll
    for (int j = 0; j < 8; ++j) {
      float p = __expf(s_buf[r][c0 + j] - mnew);
      psum += p;
      p_buf[r][c0 + j] = (__bf16)p;
    }
    red[r][t & 15] = psum;
    __syncthreads();
    if (t < 16) {
      float s_ = 0.0f;
#pragma unroll
      for (int j = 0; j < 16; ++j) s_ += red[t][j];
      lrow[t] = lrow[t] * corr[t] + s_;
    }
    __syncthreads();

    // ---- rescale O, accumulate P @ V (wave w -> d cols [w*128, +128)) ----
    float cf[8];
#pragma unroll
    for (int e = 0; e < 8; ++e) cf[e] = corr[e + 8 * hh];
#pragma unroll
    for (int nt = 0; nt < 8; ++nt)
#pragma unroll
      for (int e = 0; e < 8; ++e) o_acc[nt][e] *= cf[e];

#pragma unroll
    for (int kk = 0; kk < 4; ++kk) {
      Frag16 ap;
      const __bf16* pp = &p_buf[lm][kk * 32 + 8 * hh];
      ap.q[0] = *(const u32x4*)pp;
      ap.q[1] = *(const u32x4*)(pp + 16);
#pragma unroll
      for (int nt = 0; nt < 8; ++nt) {
        const int dcol = w * 128 + nt * 16 + lm;
        const __bf16* vp = Vt + ((size_t)b * D_MODEL + dcol) * SEQ + kv0 +
                           kk * 32 + 8 * hh;  // two 16B chunks along kv (+0, +16)
        Frag16 bvf;
        bvf.q[0] = *(const u32x4*)vp;
        bvf.q[1] = *(const u32x4*)(vp + 16);
        o_acc[nt] = wmma_bf16(ap.v, bvf.v, o_acc[nt]);
      }
    }
    __syncthreads();
  }

  // ---- finalize: O /= l, store bf16 ----
  if (t < 16) corr[t] = 1.0f / lrow[t];
  __syncthreads();
  float nf[8];
#pragma unroll
  for (int e = 0; e < 8; ++e) nf[e] = corr[e + 8 * hh];
#pragma unroll
  for (int nt = 0; nt < 8; ++nt) {
    const int dcol = w * 128 + nt * 16 + lm;
#pragma unroll
    for (int e = 0; e < 8; ++e) {
      AO[((size_t)(b * SEQ + q0 + e + 8 * hh)) * D_MODEL + dcol] =
          (__bf16)(o_acc[nt][e] * nf[e]);
    }
  }
}

// ---------------------------------------------------------------------------
extern "C" void kernel_launch(void* const* d_in, const int* in_sizes, int n_in,
                              void* d_out, int out_size, void* d_ws,
                              size_t ws_size, hipStream_t stream) {
  const float* q  = (const float*)d_in[0];
  const float* k  = (const float*)d_in[1];
  const float* v  = (const float*)d_in[2];
  const float* Wq = (const float*)d_in[3];
  const float* bq = (const float*)d_in[4];
  const float* Wk = (const float*)d_in[5];
  const float* bk = (const float*)d_in[6];
  const float* Wv = (const float*)d_in[7];
  const float* bv = (const float*)d_in[8];
  const float* Wo = (const float*)d_in[9];
  const float* bo = (const float*)d_in[10];
  float* out = (float*)d_out;

  const size_t ELEMS = (size_t)M_TOTAL * D_MODEL;  // 33.5M bf16 per buffer
  __bf16* Qp = (__bf16*)d_ws;
  __bf16* Kp = Qp + ELEMS;
  __bf16* Vt = Kp + ELEMS;
  __bf16* AO = Vt + ELEMS;

  dim3 gg(D_MODEL / 128, M_TOTAL / 128);  // (8, 256)
  const float qscale = 0.03125f;          // 1/sqrt(1024)

  gemm_proj<float, 0><<<gg, 256, 0, stream>>>(q, Wq, bq, qscale, Qp);
  gemm_proj<float, 0><<<gg, 256, 0, stream>>>(k, Wk, bk, 1.0f, Kp);
  gemm_proj<float, 1><<<gg, 256, 0, stream>>>(v, Wv, bv, 1.0f, Vt);

  attn_fused<<<dim3(BATCH * (SEQ / 16)), 256, 0, stream>>>(Qp, Kp, Vt, AO);

  gemm_proj<__bf16, 2><<<gg, 256, 0, stream>>>(AO, Wo, bo, 1.0f, out);
}
